// StackedModel_53025666236551
// MI455X (gfx1250) — matrix operand order repaired and checked by reading, compiled, not run on hardware
//
#include <hip/hip_runtime.h>

// ---------------- problem constants ----------------
constexpr int LL = 2;     // layers
constexpr int EE = 8;     // experts
constexpr int KK = 2;     // top-k
constexpr int HH = 1024;  // hidden
constexpr int II = 2048;  // intermediate
constexpr int TT = 2048;  // tokens (B*S)

// expert segments aligned to 32 rows so each 32-row WMMA super-tile is single-expert
constexpr int ROWCAP  = 4352;          // >= 4096 + 8*31, rounded to 32
constexpr int TILES32 = ROWCAP / 32;   // 136

// ---------------- WMMA fragment types ----------------
typedef __attribute__((ext_vector_type(16))) __bf16 v16bf;
typedef __attribute__((ext_vector_type(8)))  float  v8f;

union Frag {
    v16bf v;
    uint4 q[2];
};

__device__ __forceinline__ unsigned short f2bf(float f) {
    unsigned int u = __float_as_uint(f);
    unsigned int r = (u + 0x7FFFu + ((u >> 16) & 1u)) >> 16;
    return (unsigned short)r;
}

// ---------------- init: copy hidden -> xf (f32) and xb (bf16) ----------------
__global__ void k_init_x(const float* __restrict__ hs, float* __restrict__ xf,
                         unsigned short* __restrict__ xb, int n) {
    int i = blockIdx.x * blockDim.x + threadIdx.x;
    if (i < n) {
        float v = hs[i];
        xf[i] = v;
        xb[i] = f2bf(v);
    }
}

__global__ void k_convert_x(const float* __restrict__ xf,
                            unsigned short* __restrict__ xb, int n) {
    int i = blockIdx.x * blockDim.x + threadIdx.x;
    if (i < n) xb[i] = f2bf(xf[i]);
}

// ---------------- weight transpose + bf16 convert ----------------
// gate_up layer slice: [E][H][2I] fp32 -> wgt [E][2I][H] bf16 (N-major)
__global__ void k_tr_gu(const float* __restrict__ gu, unsigned short* __restrict__ wt) {
    long long idx = (long long)blockIdx.x * blockDim.x + threadIdx.x;
    const long long NI = 2LL * II;
    const long long tot = (long long)EE * HH * NI;
    if (idx >= tot) return;
    int i = (int)(idx % NI);
    long long r = idx / NI;
    int h = (int)(r % HH);
    int e = (int)(r / HH);
    float v = gu[((long long)e * HH + h) * NI + i];          // coalesced read over i
    wt[((long long)e * NI + i) * HH + h] = f2bf(v);          // strided write
}

// down layer slice: [E][I][H] fp32 -> wdt [E][H][I] bf16 (N-major)
__global__ void k_tr_dn(const float* __restrict__ dn, unsigned short* __restrict__ wt) {
    long long idx = (long long)blockIdx.x * blockDim.x + threadIdx.x;
    const long long tot = (long long)EE * II * HH;
    if (idx >= tot) return;
    int h = (int)(idx % HH);
    long long r = idx / HH;
    int i = (int)(r % II);
    int e = (int)(r / II);
    float v = dn[((long long)e * II + i) * HH + h];          // coalesced read over h
    wt[((long long)e * HH + h) * II + i] = f2bf(v);          // strided write
}

// ---------------- router: logits, softmax, top-2, normalized weights ----------------
__global__ void k_router(const float* __restrict__ xf, const float* __restrict__ rw,
                         float* __restrict__ logits_out, int* __restrict__ tokexp,
                         float* __restrict__ tokw) {
    int t = blockIdx.x * blockDim.x + threadIdx.x;
    if (t >= TT) return;
    float acc[EE];
#pragma unroll
    for (int e = 0; e < EE; ++e) acc[e] = 0.f;
    const float* xr = xf + (long long)t * HH;
    for (int h = 0; h < HH; ++h) {
        float xv = xr[h];
        const float* w = rw + h * EE;  // broadcast across lanes
#pragma unroll
        for (int e = 0; e < EE; ++e) acc[e] += xv * w[e];
    }
#pragma unroll
    for (int e = 0; e < EE; ++e) logits_out[(long long)t * EE + e] = acc[e];
    float m = acc[0];
#pragma unroll
    for (int e = 1; e < EE; ++e) m = fmaxf(m, acc[e]);
    float p[EE];
#pragma unroll
    for (int e = 0; e < EE; ++e) p[e] = __expf(acc[e] - m);
    int i0 = 0;
#pragma unroll
    for (int e = 1; e < EE; ++e) if (p[e] > p[i0]) i0 = e;
    int i1 = (i0 == 0) ? 1 : 0;
#pragma unroll
    for (int e = 0; e < EE; ++e) if (e != i0 && p[e] > p[i1]) i1 = e;
    float w0 = p[i0], w1 = p[i1];
    float inv = 1.f / (w0 + w1);   // softmax denom cancels in top-k renorm
    tokexp[t * 2 + 0] = i0; tokexp[t * 2 + 1] = i1;
    tokw[t * 2 + 0] = w0 * inv; tokw[t * 2 + 1] = w1 * inv;
}

// ---------------- expert bucketing ----------------
__global__ void k_zero(int* counts, int* cursor) {
    int i = threadIdx.x;
    if (i < EE) { counts[i] = 0; cursor[i] = 0; }
}

__global__ void k_count(const int* __restrict__ tokexp, int* __restrict__ counts) {
    int t = blockIdx.x * blockDim.x + threadIdx.x;
    if (t >= TT) return;
    atomicAdd(&counts[tokexp[t * 2 + 0]], 1);
    atomicAdd(&counts[tokexp[t * 2 + 1]], 1);
}

__global__ void k_scan(const int* __restrict__ counts, int* __restrict__ offs,
                       int* __restrict__ tilexp, int* __restrict__ rowtok) {
    int off = 0, tile = 0;
    for (int e = 0; e < EE; ++e) {
        offs[e] = off;
        int nt = (counts[e] + 31) >> 5;      // 32-row super-tiles
        for (int j = 0; j < nt; ++j) tilexp[tile++] = e;
        off += nt * 32;
    }
    offs[EE] = off;
    for (; tile < TILES32; ++tile) tilexp[tile] = -1;
    for (int r = 0; r < ROWCAP; ++r) rowtok[r] = -1;
}

__global__ void k_place(const int* __restrict__ tokexp, const int* __restrict__ offs,
                        int* __restrict__ cursor, int* __restrict__ rowtok,
                        int* __restrict__ tokrow) {
    int t = blockIdx.x * blockDim.x + threadIdx.x;
    if (t >= TT) return;
#pragma unroll
    for (int k = 0; k < KK; ++k) {
        int e = tokexp[t * 2 + k];
        int p = atomicAdd(&cursor[e], 1);
        int r = offs[e] + p;
        rowtok[r] = t;
        tokrow[t * 2 + k] = r;
    }
}

// ---------------- pass A: gate_up GEMM + SiLU*up -> act (bf16) ----------------
// grid: (II/128, TILES32), block 256 (8 waves).
// Each wave: 32 rows x 16 i-cols; A0/A1 shared by gate and up B fragments:
// 4 WMMAs per 4 fragment loads (32 B/lane per WMMA).
__global__ __launch_bounds__(256) void k_gateup(const unsigned short* __restrict__ xb,
                                                const unsigned short* __restrict__ wgt,
                                                const int* __restrict__ tilexp,
                                                const int* __restrict__ rowtok,
                                                unsigned short* __restrict__ act) {
    int tile = blockIdx.y;
    int e = tilexp[tile];
    if (e < 0) return;                       // block-uniform exit
    int lane = threadIdx.x & 31;
    int wv   = threadIdx.x >> 5;
    int n    = lane & 15;
    int half = lane >> 4;
    int colbase = blockIdx.x * 128 + wv * 16;    // i-column base
    int rowbase = tile * 32;

    int tok0 = rowtok[rowbase + n];       if (tok0 < 0) tok0 = 0;
    int tok1 = rowtok[rowbase + 16 + n];  if (tok1 < 0) tok1 = 0;
    const unsigned short* arow0 = xb + (long long)tok0 * HH;
    const unsigned short* arow1 = xb + (long long)tok1 * HH;
    const unsigned short* bg = wgt + ((long long)e * 2 * II + (colbase + n)) * HH;
    const unsigned short* bu = wgt + ((long long)e * 2 * II + II + (colbase + n)) * HH;

    v8f cg0 = {}, cg1 = {}, cu0 = {}, cu1 = {};
    for (int k0 = 0; k0 < HH; k0 += 32) {
        Frag A0, A1, Bg, Bu;
        A0.q[0] = *(const uint4*)(arow0 + k0 + half * 8);
        A0.q[1] = *(const uint4*)(arow0 + k0 + 16 + half * 8);
        A1.q[0] = *(const uint4*)(arow1 + k0 + half * 8);
        A1.q[1] = *(const uint4*)(arow1 + k0 + 16 + half * 8);
        Bg.q[0] = *(const uint4*)(bg + k0 + half * 8);
        Bg.q[1] = *(const uint4*)(bg + k0 + 16 + half * 8);
        Bu.q[0] = *(const uint4*)(bu + k0 + half * 8);
        Bu.q[1] = *(const uint4*)(bu + k0 + 16 + half * 8);
        __builtin_prefetch(bg + k0 + 128, 0, 1);
        __builtin_prefetch(bu + k0 + 128, 0, 1);
        cg0 = __builtin_amdgcn_wmma_f32_16x16x32_bf16(false, A0.v, false, Bg.v,
                                                      (short)0, cg0, false, false);
        cg1 = __builtin_amdgcn_wmma_f32_16x16x32_bf16(false, A1.v, false, Bg.v,
                                                      (short)0, cg1, false, false);
        cu0 = __builtin_amdgcn_wmma_f32_16x16x32_bf16(false, A0.v, false, Bu.v,
                                                      (short)0, cu0, false, false);
        cu1 = __builtin_amdgcn_wmma_f32_16x16x32_bf16(false, A1.v, false, Bu.v,
                                                      (short)0, cu1, false, false);
    }
    long long col = colbase + n;
#pragma unroll
    for (int v = 0; v < 8; ++v) {
        int mr = half * 8 + v;
        float g0 = cg0[v], u0 = cu0[v];
        float g1 = cg1[v], u1 = cu1[v];
        float s0 = (g0 / (1.f + __expf(-g0))) * u0;   // SiLU(gate) * up
        float s1 = (g1 / (1.f + __expf(-g1))) * u1;
        act[(long long)(rowbase + mr) * II + col]      = f2bf(s0);
        act[(long long)(rowbase + 16 + mr) * II + col] = f2bf(s1);
    }
}

// ---------------- pass B: down GEMM -> part (f32) ----------------
// grid: (HH/256, TILES32), block 256 (8 waves).
// Each wave: 32 rows x 32 h-cols; A0/A1 shared by B0/B1:
// 4 WMMAs per 4 fragment loads (32 B/lane per WMMA).
__global__ __launch_bounds__(256) void k_down(const unsigned short* __restrict__ act,
                                              const unsigned short* __restrict__ wdt,
                                              const int* __restrict__ tilexp,
                                              float* __restrict__ part) {
    int tile = blockIdx.y;
    int e = tilexp[tile];
    if (e < 0) return;
    int lane = threadIdx.x & 31;
    int wv   = threadIdx.x >> 5;
    int n    = lane & 15;
    int half = lane >> 4;
    int colbase = blockIdx.x * 256 + wv * 32;    // h-column base
    int rowbase = tile * 32;

    const unsigned short* arow0 = act + (long long)(rowbase + n) * II;
    const unsigned short* arow1 = act + (long long)(rowbase + 16 + n) * II;
    const unsigned short* bd0 = wdt + ((long long)e * HH + (colbase + n)) * II;
    const unsigned short* bd1 = wdt + ((long long)e * HH + (colbase + 16 + n)) * II;

    v8f c00 = {}, c01 = {}, c10 = {}, c11 = {};
    for (int k0 = 0; k0 < II; k0 += 32) {
        Frag A0, A1, B0, B1;
        A0.q[0] = *(const uint4*)(arow0 + k0 + half * 8);
        A0.q[1] = *(const uint4*)(arow0 + k0 + 16 + half * 8);
        A1.q[0] = *(const uint4*)(arow1 + k0 + half * 8);
        A1.q[1] = *(const uint4*)(arow1 + k0 + 16 + half * 8);
        B0.q[0] = *(const uint4*)(bd0 + k0 + half * 8);
        B0.q[1] = *(const uint4*)(bd0 + k0 + 16 + half * 8);
        B1.q[0] = *(const uint4*)(bd1 + k0 + half * 8);
        B1.q[1] = *(const uint4*)(bd1 + k0 + 16 + half * 8);
        __builtin_prefetch(bd0 + k0 + 128, 0, 1);
        __builtin_prefetch(bd1 + k0 + 128, 0, 1);
        c00 = __builtin_amdgcn_wmma_f32_16x16x32_bf16(false, A0.v, false, B0.v,
                                                      (short)0, c00, false, false);
        c10 = __builtin_amdgcn_wmma_f32_16x16x32_bf16(false, A1.v, false, B0.v,
                                                      (short)0, c10, false, false);
        c01 = __builtin_amdgcn_wmma_f32_16x16x32_bf16(false, A0.v, false, B1.v,
                                                      (short)0, c01, false, false);
        c11 = __builtin_amdgcn_wmma_f32_16x16x32_bf16(false, A1.v, false, B1.v,
                                                      (short)0, c11, false, false);
    }
    long long col0 = colbase + n;
    long long col1 = colbase + 16 + n;
#pragma unroll
    for (int v = 0; v < 8; ++v) {
        int mr = half * 8 + v;
        part[(long long)(rowbase + mr) * HH + col0]      = c00[v];
        part[(long long)(rowbase + mr) * HH + col1]      = c01[v];
        part[(long long)(rowbase + 16 + mr) * HH + col0] = c10[v];
        part[(long long)(rowbase + 16 + mr) * HH + col1] = c11[v];
    }
}

// ---------------- combine: deterministic 2-way weighted sum per token ----------------
__global__ void k_combine(const float* __restrict__ part, const int* __restrict__ tokrow,
                          const float* __restrict__ tokw, float* __restrict__ xf,
                          float* __restrict__ dout, int final_layer) {
    int t = blockIdx.x;
    int r0 = tokrow[t * 2 + 0], r1 = tokrow[t * 2 + 1];
    float w0 = tokw[t * 2 + 0], w1 = tokw[t * 2 + 1];
    const float* p0 = part + (long long)r0 * HH;
    const float* p1 = part + (long long)r1 * HH;
    for (int h = threadIdx.x; h < HH; h += blockDim.x) {
        float v = w0 * p0[h] + w1 * p1[h];
        xf[(long long)t * HH + h] = v;
        if (final_layer) dout[(long long)t * HH + h] = v;
    }
}

// ---------------- host launcher ----------------
extern "C" void kernel_launch(void* const* d_in, const int* in_sizes, int n_in,
                              void* d_out, int out_size, void* d_ws, size_t ws_size,
                              hipStream_t stream) {
    const float* hs = (const float*)d_in[0];   // [B,S,H]
    const float* rw = (const float*)d_in[1];   // [L,H,E]
    const float* gu = (const float*)d_in[2];   // [L,E,H,2I]
    const float* dn = (const float*)d_in[3];   // [L,E,I,H]
    float* out = (float*)d_out;                // [T*H] out, then [L*T*E] logits

    char* ws = (char*)d_ws;
    size_t off = 0;
    auto alloc = [&](size_t bytes) -> char* {
        off = (off + 255) & ~(size_t)255;
        char* p = ws + off;
        off += bytes;
        return p;
    };
    float*          xf     = (float*)alloc((size_t)TT * HH * 4);
    unsigned short* xb     = (unsigned short*)alloc((size_t)TT * HH * 2);
    unsigned short* wgt    = (unsigned short*)alloc((size_t)EE * 2 * II * HH * 2);
    unsigned short* wdt    = (unsigned short*)alloc((size_t)EE * HH * II * 2);
    unsigned short* act    = (unsigned short*)alloc((size_t)ROWCAP * II * 2);
    float*          part   = (float*)alloc((size_t)ROWCAP * HH * 4);
    int*            counts = (int*)alloc(EE * 4);
    int*            cursor = (int*)alloc(EE * 4);
    int*            offs   = (int*)alloc((EE + 1) * 4);
    int*            tilexp = (int*)alloc(TILES32 * 4);
    int*            rowtok = (int*)alloc(ROWCAP * 4);
    int*            tokexp = (int*)alloc((size_t)TT * 2 * 4);
    int*            tokrow = (int*)alloc((size_t)TT * 2 * 4);
    float*          tokw   = (float*)alloc((size_t)TT * 2 * 4);

    const int nXH = TT * HH;
    k_init_x<<<(nXH + 255) / 256, 256, 0, stream>>>(hs, xf, xb, nXH);

    for (int l = 0; l < LL; ++l) {
        const float* gu_l = gu + (size_t)l * EE * HH * 2 * II;
        const float* dn_l = dn + (size_t)l * EE * II * HH;
        const float* rw_l = rw + (size_t)l * HH * EE;
        float* logits_l = out + (size_t)TT * HH + (size_t)l * TT * EE;

        long long ngu = (long long)EE * HH * 2 * II;
        long long ndn = (long long)EE * II * HH;
        k_tr_gu<<<(unsigned)((ngu + 255) / 256), 256, 0, stream>>>(gu_l, wgt);
        k_tr_dn<<<(unsigned)((ndn + 255) / 256), 256, 0, stream>>>(dn_l, wdt);

        k_router<<<(TT + 255) / 256, 256, 0, stream>>>(xf, rw_l, logits_l, tokexp, tokw);
        k_zero<<<1, 32, 0, stream>>>(counts, cursor);
        k_count<<<(TT + 255) / 256, 256, 0, stream>>>(tokexp, counts);
        k_scan<<<1, 1, 0, stream>>>(counts, offs, tilexp, rowtok);
        k_place<<<(TT + 255) / 256, 256, 0, stream>>>(tokexp, offs, cursor, rowtok, tokrow);

        k_gateup<<<dim3(II / 128, TILES32), 256, 0, stream>>>(xb, wgt, tilexp, rowtok, act);
        k_down<<<dim3(HH / 256, TILES32), 256, 0, stream>>>(act, wdt, tilexp, part);

        k_combine<<<TT, 256, 0, stream>>>(part, tokrow, tokw, xf, out, (l == LL - 1) ? 1 : 0);
        if (l != LL - 1) {
            k_convert_x<<<(nXH + 255) / 256, 256, 0, stream>>>(xf, xb, nXH);
        }
    }
}
